// RecurrentCellsModel_26285199852090
// MI455X (gfx1250) — matrix operand order repaired and checked by reading, not compiled, not run on hardware
//
#include <hip/hip_runtime.h>
#include <hip/hip_bf16.h>

typedef __attribute__((ext_vector_type(16))) _Float16 v16h;
typedef __attribute__((ext_vector_type(8)))  _Float16 v8h;
typedef __attribute__((ext_vector_type(8)))  float    v8f;

#define B_SZ   512
#define T_LEN  2048
#define VOCAB  128
#define D_IN   32
#define HIDN   48
#define OUT_D  6

// ---- LDS layout (bytes). Fragment image = 32 lanes * 16 halfs = 1024 B ----
#define OFF_WIH_G  0          //  9 frags  (gru_wih : N=144, K=32,  1 K-chunk)
#define OFF_WHH_G  9216       // 18 frags  (gru_whh : N=144, K=48->64, 2 K-chunks)
#define OFF_WIH_L  27648      // 24 frags  (lstm_wih: N=192, K=48->64)
#define OFF_WHH_L  52224      // 24 frags  (lstm_whh: N=192, K=48->64)
#define OFF_EMB    76800      // 128*32 f16 = 8192 B
#define OFF_HGA    84992      // hg state, f16 [16][64] = 2048 B (cols 48..63 stay 0)
#define OFF_HA     87040      // h  state, f16 [16][64] = 2048 B
#define OFF_HOUT   89088      // final h, f32 [16][48] = 3072 B
#define SMEM_BYTES 92160

// Fast activations: raw v_exp_f32 + v_rcp_f32 (no IEEE div expansion).
__device__ __forceinline__ float frcp(float x) { return __builtin_amdgcn_rcpf(x); }
__device__ __forceinline__ float sigf(float x) { return frcp(1.f + __expf(-x)); }
__device__ __forceinline__ float tanhfast(float x) {
  float e = __expf(2.f * x);
  return (e - 1.f) * frcp(e + 1.f);
}

// Build a v16h from two contiguous 8-half (16B) chunks -> two ds_load_b128.
__device__ __forceinline__ v16h pack16(const _Float16* p0, const _Float16* p1) {
  v8h lo = *(const v8h*)p0;
  v8h hi = *(const v8h*)p1;
  v16h a;
#pragma unroll
  for (int j = 0; j < 8; ++j) { a[j] = lo[j]; a[j + 8] = hi[j]; }
  return a;
}

// B fragment image: lane ln owns 16 consecutive halfs at fbase + ln*16.
__device__ __forceinline__ v16h load_frag(const _Float16* fbase, int ln) {
  return pack16(fbase + ln * 16, fbase + ln * 16 + 8);
}

__device__ __forceinline__ v8f wmma_f16(v16h a, v16h b, v8f c) {
  // D = A(16x32 f16) * B(32x16 f16) + C(16x16 f32)
  return __builtin_amdgcn_wmma_f32_16x16x32_f16(false, a, false, b, (short)0, c,
                                                false, false);
}

// Pre-swizzle one B fragment for x @ W^T:  B[k][n] = W[n][k].
// Layout (16-bit B, 32x16): lanes 0-15 -> n=lane, K=kc*32+{0..7,16..23};
//                           lanes 16-31 -> n=lane-16, K=kc*32+{8..15,24..31}.
__device__ __forceinline__ void fill_B_frag(_Float16* dst, const float* W,
                                            int Nout, int Kin, int ntile, int kc,
                                            int ln) {
  int n  = ntile * 16 + (ln & 15);
  int kb = kc * 32 + ((ln < 16) ? 0 : 8);
#pragma unroll
  for (int j = 0; j < 16; ++j) {
    int k = kb + (j < 8 ? j : j + 8);
    float v = (n < Nout && k < Kin) ? W[n * Kin + k] : 0.f;
    dst[ln * 16 + j] = (_Float16)v;
  }
}

__global__ __launch_bounds__(32) void rnn_wmma_kernel(
    const int* __restrict__ token_ids, const float* __restrict__ emb,
    const float* __restrict__ gru_wih, const float* __restrict__ gru_whh,
    const float* __restrict__ gru_b, const float* __restrict__ gru_bn,
    const float* __restrict__ lstm_wih, const float* __restrict__ lstm_whh,
    const float* __restrict__ lstm_b, const float* __restrict__ head_w,
    const float* __restrict__ head_b, float* __restrict__ out) {
  extern __shared__ __align__(16) char smem[];
  _Float16* wih_g = (_Float16*)(smem + OFF_WIH_G);
  _Float16* whh_g = (_Float16*)(smem + OFF_WHH_G);
  _Float16* wih_l = (_Float16*)(smem + OFF_WIH_L);
  _Float16* whh_l = (_Float16*)(smem + OFF_WHH_L);
  _Float16* embh  = (_Float16*)(smem + OFF_EMB);
  _Float16* hgA   = (_Float16*)(smem + OFF_HGA);
  _Float16* hA    = (_Float16*)(smem + OFF_HA);
  float*    hout  = (float*)(smem + OFF_HOUT);

  const int ln    = threadIdx.x;        // single wave32 per block
  const int nl    = ln & 15;            // N column / batch row in half-wave
  const int mbase = (ln < 16) ? 0 : 8;  // C/D layout: lanes 16-31 hold M=r+8
  const int kh    = (ln < 16) ? 0 : 8;  // A layout K-half offset
  const int b0    = blockIdx.x * 16;    // this wave's batch tile

  // -------- one-time setup: weights -> f16 fragment images, emb -> f16 ------
  for (int nt = 0; nt < 9; ++nt)
    fill_B_frag(wih_g + nt * 512, gru_wih, 3 * HIDN, D_IN, nt, 0, ln);
  for (int nt = 0; nt < 9; ++nt)
    for (int kc = 0; kc < 2; ++kc)
      fill_B_frag(whh_g + (nt * 2 + kc) * 512, gru_whh, 3 * HIDN, HIDN, nt, kc, ln);
  for (int nt = 0; nt < 12; ++nt)
    for (int kc = 0; kc < 2; ++kc) {
      fill_B_frag(wih_l + (nt * 2 + kc) * 512, lstm_wih, 4 * HIDN, HIDN, nt, kc, ln);
      fill_B_frag(whh_l + (nt * 2 + kc) * 512, lstm_whh, 4 * HIDN, HIDN, nt, kc, ln);
    }
  for (int i = ln; i < VOCAB * D_IN; i += 32) embh[i] = (_Float16)emb[i];
  for (int i = ln; i < 16 * 64; i += 32) { hgA[i] = (_Float16)0.f; hA[i] = (_Float16)0.f; }

  // per-lane bias scalars (bias depends only on gate column n = nl)
  float gb[9], bnl[3], lb[12];
#pragma unroll
  for (int tt = 0; tt < 9; ++tt) gb[tt] = gru_b[tt * 16 + nl];
#pragma unroll
  for (int tt = 0; tt < 3; ++tt) bnl[tt] = gru_bn[tt * 16 + nl];
#pragma unroll
  for (int tt = 0; tt < 12; ++tt) lb[tt] = lstm_b[tt * 16 + nl];

  // recurrent state kept in registers in WMMA C/D layout (3 N-tiles each)
  v8f hg_s[3], h_s[3], c_s[3];
#pragma unroll
  for (int tt = 0; tt < 3; ++tt)
#pragma unroll
    for (int j = 0; j < 8; ++j) { hg_s[tt][j] = 0.f; h_s[tt][j] = 0.f; c_s[tt][j] = 0.f; }

  const int* tokrow = token_ids + (size_t)(b0 + nl) * T_LEN;
  const _Float16* hgrow = hgA + nl * 64;
  const _Float16* hrow  = hA + nl * 64;

  asm volatile("s_wait_dscnt 0" ::: "memory");

  for (int t = 0; t < T_LEN; ++t) {
    int tp = t + 64; tp = (tp < T_LEN) ? tp : (T_LEN - 1);
    __builtin_prefetch(tokrow + tp, 0, 1);

    // ---- A fragments: x_t (from f16 embedding) and hg_old (from LDS) ----
    int tok = tokrow[t];
    const _Float16* ep = embh + tok * 32;
    v16h a_x   = pack16(ep + kh, ep + 16 + kh);
    v16h a_hg0 = pack16(hgrow + kh, hgrow + 16 + kh);        // K = 0..31
    v16h a_hg1 = pack16(hgrow + 32 + kh, hgrow + 48 + kh);   // K = 32..63

    // ---- GRU, one hidden-chunk (16 cols) at a time: 4 live accumulators ----
#pragma unroll
    for (int tt = 0; tt < 3; ++tt) {
      v8f zero = {};
      // r-gate tile tt : ig + hgates accumulated together
      v8f ar = wmma_f16(a_x, load_frag(wih_g + tt * 512, ln), zero);
      ar = wmma_f16(a_hg0, load_frag(whh_g + (tt * 2 + 0) * 512, ln), ar);
      ar = wmma_f16(a_hg1, load_frag(whh_g + (tt * 2 + 1) * 512, ln), ar);
      // z-gate tile tt+3
      v8f az = wmma_f16(a_x, load_frag(wih_g + (tt + 3) * 512, ln), zero);
      az = wmma_f16(a_hg0, load_frag(whh_g + ((tt + 3) * 2 + 0) * 512, ln), az);
      az = wmma_f16(a_hg1, load_frag(whh_g + ((tt + 3) * 2 + 1) * 512, ln), az);
      // n-gate tile tt+6: ig and hgates kept split
      v8f ai = wmma_f16(a_x, load_frag(wih_g + (tt + 6) * 512, ln), zero);
      v8f ah = wmma_f16(a_hg0, load_frag(whh_g + ((tt + 6) * 2 + 0) * 512, ln), zero);
      ah = wmma_f16(a_hg1, load_frag(whh_g + ((tt + 6) * 2 + 1) * 512, ln), ah);

#pragma unroll
      for (int j = 0; j < 8; ++j) {
        float rv  = sigf(ar[j] + gb[tt]);
        float zv  = sigf(az[j] + gb[tt + 3]);
        float nv  = tanhfast(ai[j] + gb[tt + 6] + rv * (ah[j] + bnl[tt]));
        float hgv = nv + zv * (hg_s[tt][j] - nv);
        hg_s[tt][j] = hgv;
        hgA[(j + mbase) * 64 + tt * 16 + nl] = (_Float16)hgv;
      }
    }
    asm volatile("s_wait_dscnt 0" ::: "memory");

    // ---- LSTM: lin = hg_new @ Wih^T + h @ Whh^T + b, chunk-at-a-time ----
    v16h a_g0 = pack16(hgrow + kh, hgrow + 16 + kh);
    v16h a_g1 = pack16(hgrow + 32 + kh, hgrow + 48 + kh);
    v16h a_h0 = pack16(hrow + kh, hrow + 16 + kh);
    v16h a_h1 = pack16(hrow + 32 + kh, hrow + 48 + kh);

#pragma unroll
    for (int tt = 0; tt < 3; ++tt) {
      v8f gate[4];  // i (tt), f (tt+3), g (tt+6), o (tt+9)
#pragma unroll
      for (int q = 0; q < 4; ++q) {
        int tile = tt + 3 * q;
        v8f c = {};
        c = wmma_f16(a_g0, load_frag(wih_l + (tile * 2 + 0) * 512, ln), c);
        c = wmma_f16(a_g1, load_frag(wih_l + (tile * 2 + 1) * 512, ln), c);
        c = wmma_f16(a_h0, load_frag(whh_l + (tile * 2 + 0) * 512, ln), c);
        c = wmma_f16(a_h1, load_frag(whh_l + (tile * 2 + 1) * 512, ln), c);
        gate[q] = c;
      }
#pragma unroll
      for (int j = 0; j < 8; ++j) {
        float iv = sigf(gate[0][j] + lb[tt]);
        float fv = sigf(gate[1][j] + lb[tt + 3]);
        float gv = tanhfast(gate[2][j] + lb[tt + 6]);
        float ov = sigf(gate[3][j] + lb[tt + 9]);
        float cv = fv * c_s[tt][j] + iv * gv;
        float hv = ov * tanhfast(cv);
        c_s[tt][j] = cv;
        h_s[tt][j] = hv;
        hA[(j + mbase) * 64 + tt * 16 + nl] = (_Float16)hv;
      }
    }
    asm volatile("s_wait_dscnt 0" ::: "memory");
  }

  // -------- head: out = h @ head_w^T + head_b  ([16,48]x[48,6], tiny) -----
#pragma unroll
  for (int tt = 0; tt < 3; ++tt)
#pragma unroll
    for (int j = 0; j < 8; ++j)
      hout[(j + mbase) * 48 + tt * 16 + nl] = h_s[tt][j];
  asm volatile("s_wait_dscnt 0" ::: "memory");

  if (ln < 16) {
    int m = ln;
#pragma unroll
    for (int o = 0; o < OUT_D; ++o) {
      float s = head_b[o];
      for (int k = 0; k < HIDN; ++k) s += hout[m * 48 + k] * head_w[o * 48 + k];
      out[(size_t)(b0 + m) * OUT_D + o] = s;
    }
  }
}

extern "C" void kernel_launch(void* const* d_in, const int* in_sizes, int n_in,
                              void* d_out, int out_size, void* d_ws, size_t ws_size,
                              hipStream_t stream) {
  (void)in_sizes; (void)n_in; (void)d_ws; (void)ws_size; (void)out_size;
  const int*   token_ids = (const int*)d_in[0];
  const float* emb       = (const float*)d_in[1];
  const float* gru_wih   = (const float*)d_in[2];
  const float* gru_whh   = (const float*)d_in[3];
  const float* gru_b     = (const float*)d_in[4];
  const float* gru_bn    = (const float*)d_in[5];
  const float* lstm_wih  = (const float*)d_in[6];
  const float* lstm_whh  = (const float*)d_in[7];
  const float* lstm_b    = (const float*)d_in[8];
  const float* head_w    = (const float*)d_in[9];
  const float* head_b    = (const float*)d_in[10];
  float* out = (float*)d_out;

  // > 64 KB dynamic LDS: raise the per-kernel cap (host-side attr, capture-safe)
  hipFuncSetAttribute((const void*)rnn_wmma_kernel,
                      hipFuncAttributeMaxDynamicSharedMemorySize, SMEM_BYTES);

  dim3 grid(B_SZ / 16), block(32);  // 32 independent single-wave workgroups
  hipLaunchKernelGGL(rnn_wmma_kernel, grid, block, SMEM_BYTES, stream,
                     token_ids, emb, gru_wih, gru_whh, gru_b, gru_bn,
                     lstm_wih, lstm_whh, lstm_b, head_w, head_b, out);
}
